// VectorQuantizerEMA_70059506532353
// MI455X (gfx1250) — matrix operand order repaired and checked by reading, not compiled, not run on hardware
//
#include <hip/hip_runtime.h>
#include <hip/hip_bf16.h>

#define N_VEC 32768   // 32*32*32
#define DIM   256
#define KCODE 1024
#define BETA  0.25f
#define KT    32      // codes per LDS chunk
#define NCHUNK (KCODE / KT)   // 32

typedef __attribute__((ext_vector_type(16))) __bf16        v16bf;
typedef __attribute__((ext_vector_type(8)))  float         v8f;
typedef __attribute__((ext_vector_type(4)))  unsigned int  v4u;
typedef __attribute__((ext_vector_type(8)))  int           v8i;
typedef __attribute__((ext_vector_type(4)))  int           v4i;

union FragA { unsigned int u[8]; v16bf v; };

static __device__ __forceinline__ unsigned short f32_to_bf16_rne(float f) {
    unsigned int u = __float_as_uint(f);
    unsigned int r = 0x7FFFu + ((u >> 16) & 1u);
    return (unsigned short)((u + r) >> 16);
}
static __device__ __forceinline__ float bf16_bits_to_f32(unsigned short h) {
    return __uint_as_float(((unsigned int)h) << 16);
}

// ---------------------------------------------------------------------------
// Tensor Data Mover: 1-D contiguous tile (n 2-byte elements) global -> LDS.
// D# layout per CDNA5 ISA.
// ---------------------------------------------------------------------------
static __device__ __forceinline__ void tdm_load_1d_2B(const void* gsrc,
                                                      unsigned int lds_byte_addr,
                                                      unsigned int n_elems) {
    unsigned long long ga = (unsigned long long)gsrc;
    v4u g0;
    g0[0] = 1u;                                         // count=1, user D#
    g0[1] = lds_byte_addr;                              // LDS dest (bytes)
    g0[2] = (unsigned int)(ga & 0xFFFFFFFFu);           // global_addr[31:0]
    g0[3] = (unsigned int)((ga >> 32) & 0x01FFFFFFu)    // global_addr[56:32]
          | (2u << 30);                                 // type=2 ("image")
    v8i g1;
    g1[0] = (int)(1u << 16);                            // data_size=1 -> 2 bytes
    g1[1] = (int)((n_elems & 0xFFFFu) << 16);           // tensor_dim0[15:0]
    g1[2] = (int)(((n_elems >> 16) & 0xFFFFu)           // tensor_dim0[31:16]
          | (1u << 16));                                // tensor_dim1 = 1
    g1[3] = (int)((n_elems & 0xFFFFu) << 16);           // tile_dim0 = n
    g1[4] = 1;                                          // tile_dim1 = 1
    g1[5] = (int)n_elems;                               // tensor_dim0_stride[31:0]
    g1[6] = 0;
    g1[7] = 0;
    v4i gz = {0, 0, 0, 0};
#if __clang_major__ >= 23
    typedef __attribute__((ext_vector_type(8))) int v8i_;
    v8i_ gz8 = {0, 0, 0, 0, 0, 0, 0, 0};
    __builtin_amdgcn_tensor_load_to_lds(g0, g1, gz, gz, gz8, 0);
#else
    __builtin_amdgcn_tensor_load_to_lds(g0, g1, gz, gz, 0);
#endif
}

static __device__ __forceinline__ unsigned int lds_offset_of(const void* p) {
    return (unsigned int)(unsigned long long)p;   // low 32 bits = LDS byte offset
}

// ---------------------------------------------------------------------------
// Kernel 0: zero the atomic scratch (counts histogram + sse accumulator)
// ---------------------------------------------------------------------------
__global__ void vq_zero_kernel(float* __restrict__ counts, float* __restrict__ sse) {
    int t = threadIdx.x;
    for (int i = t; i < KCODE; i += 256) counts[i] = 0.0f;
    if (t == 0) *sse = 0.0f;
}

// ---------------------------------------------------------------------------
// Kernel 1: codebook prep — bf16 hi/lo split + exact f32 row norms
// ---------------------------------------------------------------------------
__global__ void vq_prep_kernel(const float* __restrict__ cb,
                               unsigned short* __restrict__ cb_hi,
                               unsigned short* __restrict__ cb_lo,
                               float* __restrict__ enorm) {
    __shared__ float s[256];
    const int k = blockIdx.x, t = threadIdx.x;
    float f = cb[k * DIM + t];
    unsigned short hi = f32_to_bf16_rne(f);
    float hif = bf16_bits_to_f32(hi);
    unsigned short lo = f32_to_bf16_rne(f - hif);
    cb_hi[k * DIM + t] = hi;
    cb_lo[k * DIM + t] = lo;
    s[t] = f * f;
    __syncthreads();
    for (int o = 128; o; o >>= 1) { if (t < o) s[t] += s[t + o]; __syncthreads(); }
    if (t == 0) enorm[k] = s[0];
}

// ---------------------------------------------------------------------------
// Kernel 2: WMMA distance + argmin.
// 8 waves / block, 16 z-rows per wave (register-resident bf16 hi/lo A).
// Codebook swept in 32-code chunks staged into LDS by the TDM (wave 0),
// double-buffered, synchronized with s_wait_tensorcnt + workgroup barriers.
// ---------------------------------------------------------------------------
__global__ void __launch_bounds__(256)
vq_argmin_kernel(const float* __restrict__ z,
                 const unsigned short* __restrict__ cb_hi,
                 const unsigned short* __restrict__ cb_lo,
                 const float* __restrict__ enorm,
                 int* __restrict__ idx_out,
                 float* __restrict__ idx_f_out) {
    __shared__ __align__(64) unsigned short lds_hi[2][KT * DIM];   // 2 x 16 KB
    __shared__ __align__(64) unsigned short lds_lo[2][KT * DIM];   // 2 x 16 KB

    const int lane = threadIdx.x & 31;
    const int wave = threadIdx.x >> 5;
    const int m0   = (blockIdx.x * 8 + wave) * 16;
    const int col  = lane & 15;      // B/C column
    const int half = lane >> 4;      // lane half selects K sub-range

    // ---- Kick off chunk 0 staging (hi + lo) via the Tensor Data Mover ----
    if (wave == 0) {
        tdm_load_1d_2B(cb_hi, lds_offset_of(&lds_hi[0][0]), KT * DIM);
        tdm_load_1d_2B(cb_lo, lds_offset_of(&lds_lo[0][0]), KT * DIM);
    }

    // ---- Load this wave's 16 z rows, convert to bf16 hi/lo A fragments ----
    // A fragments (16x32), contiguous k order (B matches): lanes 0-15 hold K = {0..15}+32c of row
    // m0+lane; lanes 16-31 hold K = {16..31}+32c of row m0+lane-16.
    FragA a_hi[8], a_lo[8];
    float zpart = 0.0f;
    const float* zr = z + (size_t)(m0 + col) * DIM;
    #pragma unroll
    for (int c = 0; c < 8; ++c) {
        const int d0 = c * 32 + half * 16;
        float f[16];
        const float4* p0 = (const float4*)(zr + d0);
        const float4* p1 = (const float4*)(zr + d0 + 8);
        float4 q0 = p0[0], q1 = p0[1], q2 = p1[0], q3 = p1[1];
        f[0]=q0.x; f[1]=q0.y; f[2]=q0.z; f[3]=q0.w;
        f[4]=q1.x; f[5]=q1.y; f[6]=q1.z; f[7]=q1.w;
        f[8]=q2.x; f[9]=q2.y; f[10]=q2.z; f[11]=q2.w;
        f[12]=q3.x; f[13]=q3.y; f[14]=q3.z; f[15]=q3.w;
        #pragma unroll
        for (int j = 0; j < 8; ++j) {
            unsigned short h0 = f32_to_bf16_rne(f[2*j]);
            unsigned short h1 = f32_to_bf16_rne(f[2*j+1]);
            unsigned short l0 = f32_to_bf16_rne(f[2*j]   - bf16_bits_to_f32(h0));
            unsigned short l1 = f32_to_bf16_rne(f[2*j+1] - bf16_bits_to_f32(h1));
            a_hi[c].u[j] = (unsigned int)h0 | ((unsigned int)h1 << 16);
            a_lo[c].u[j] = (unsigned int)l0 | ((unsigned int)l1 << 16);
            zpart += f[2*j]*f[2*j] + f[2*j+1]*f[2*j+1];
        }
    }
    // Row norm: lanes L and L+16 jointly cover all 256 elements of row m0+L%16.
    zpart += __shfl_xor(zpart, 16, 32);
    // zn[j] = ||z_row||^2 for C-row (m0 + j + 8*half): source lane = 8*half+j.
    float zn[8];
    #pragma unroll
    for (int j = 0; j < 8; ++j) zn[j] = __shfl(zpart, half * 8 + j, 32);

    // ---- Sweep codes: double-buffered TDM-staged 32-code chunks ----
    float minv[8]; int mini[8];
    #pragma unroll
    for (int j = 0; j < 8; ++j) { minv[j] = 3.402823466e+38f; mini[j] = 0; }

    for (int c = 0; c < NCHUNK; ++c) {
        __syncthreads();   // everyone done with the buffer chunk c+1 will reuse
        if (wave == 0) {
            if (c + 1 < NCHUNK) {
                const int nb = (c + 1) & 1;
                tdm_load_1d_2B(cb_hi + (size_t)(c + 1) * KT * DIM,
                               lds_offset_of(&lds_hi[nb][0]), KT * DIM);
                tdm_load_1d_2B(cb_lo + (size_t)(c + 1) * KT * DIM,
                               lds_offset_of(&lds_lo[nb][0]), KT * DIM);
                __builtin_amdgcn_s_wait_tensorcnt(2);   // chunk c complete
            } else {
                __builtin_amdgcn_s_wait_tensorcnt(0);
            }
        }
        __syncthreads();   // publish chunk c to all waves
        const int buf = c & 1;

        #pragma unroll
        for (int kt2 = 0; kt2 < KT / 16; ++kt2) {
            const int e = c * KT + kt2 * 16 + col;     // code index (this lane)
            const float en = enorm[e];
            // B fragment (32x16 bf16), k order as in A: lane holds column N=col, 16 contiguous
            // K values picked by lane half -> 32B contiguous LDS load.
            const unsigned short* bh = &lds_hi[buf][(kt2 * 16 + col) * DIM + half * 16];
            const unsigned short* bl = &lds_lo[buf][(kt2 * 16 + col) * DIM + half * 16];
            v8f acc = {};
            #pragma unroll
            for (int cc = 0; cc < 8; ++cc) {
                v16bf bhv = *(const v16bf*)(bh + cc * 32);
                v16bf blv = *(const v16bf*)(bl + cc * 32);
                // hi*hi + lo*hi + hi*lo ~= f32-accurate cross products
                acc = __builtin_amdgcn_wmma_f32_16x16x32_bf16(
                          false, a_hi[cc].v, false, bhv, (short)0, acc, false, false);
                acc = __builtin_amdgcn_wmma_f32_16x16x32_bf16(
                          false, a_lo[cc].v, false, bhv, (short)0, acc, false, false);
                acc = __builtin_amdgcn_wmma_f32_16x16x32_bf16(
                          false, a_hi[cc].v, false, blv, (short)0, acc, false, false);
            }
            #pragma unroll
            for (int j = 0; j < 8; ++j) {
                float dist = zn[j] + en - 2.0f * acc[j];
                if (dist < minv[j]) { minv[j] = dist; mini[j] = e; }
            }
        }
    }

    // ---- Cross-lane argmin within each 16-lane column group ----
    #pragma unroll
    for (int j = 0; j < 8; ++j) {
        float v = minv[j]; int i = mini[j];
        #pragma unroll
        for (int m = 8; m; m >>= 1) {
            float ov = __shfl_xor(v, m, 32);
            int   oi = __shfl_xor(i, m, 32);
            if (ov < v || (ov == v && oi < i)) { v = ov; i = oi; }
        }
        if (col == 0) {
            int r = m0 + half * 8 + j;
            idx_out[r]   = i;
            idx_f_out[r] = (float)i;
        }
    }
}

// ---------------------------------------------------------------------------
// Kernel 3: gather quantized vectors, accumulate SSE + code histogram
// ---------------------------------------------------------------------------
__global__ void vq_gather_kernel(const float* __restrict__ z,
                                 const float* __restrict__ cb,
                                 const int* __restrict__ idx,
                                 float* __restrict__ qout,
                                 float* __restrict__ counts,
                                 float* __restrict__ sse) {
    __shared__ float s[256];
    const int r = blockIdx.x, t = threadIdx.x;
    const int e = idx[r];
    float q  = cb[e * DIM + t];
    float zv = z[(size_t)r * DIM + t];
    qout[(size_t)r * DIM + t] = q;   // straight-through fwd == quantized
    float d = q - zv;
    s[t] = d * d;
    __syncthreads();
    for (int o = 128; o; o >>= 1) { if (t < o) s[t] += s[t + o]; __syncthreads(); }
    if (t == 0) {
        atomicAdd(sse, s[0]);
        atomicAdd(&counts[e], 1.0f);
    }
}

// ---------------------------------------------------------------------------
// Kernel 4: scalars — zero, commitment loss, perplexity
// ---------------------------------------------------------------------------
__global__ void vq_finalize_kernel(const float* __restrict__ counts,
                                   const float* __restrict__ sse,
                                   float* __restrict__ out_scalars) {
    __shared__ float s[256];
    const int t = threadIdx.x;
    float acc = 0.0f;
    for (int i = t; i < KCODE; i += 256) {
        float p = counts[i] * (1.0f / (float)N_VEC);
        acc += p * logf(p + 1e-10f);
    }
    s[t] = acc;
    __syncthreads();
    for (int o = 128; o; o >>= 1) { if (t < o) s[t] += s[t + o]; __syncthreads(); }
    if (t == 0) {
        out_scalars[0] = 0.0f;                                            // zeros(())
        out_scalars[1] = BETA * (*sse) * (1.0f / ((float)N_VEC * DIM));   // commitment
        out_scalars[2] = expf(-s[0]);                                     // perplexity
    }
}

// ---------------------------------------------------------------------------
extern "C" void kernel_launch(void* const* d_in, const int* in_sizes, int n_in,
                              void* d_out, int out_size, void* d_ws, size_t ws_size,
                              hipStream_t stream) {
    const float* z  = (const float*)d_in[0];   // [32,32,32,256] f32
    const float* cb = (const float*)d_in[1];   // [1024,256] f32
    float* out = (float*)d_out;

    // ws layout
    char* ws = (char*)d_ws;
    unsigned short* cb_hi = (unsigned short*)(ws);                        // 512 KB
    unsigned short* cb_lo = (unsigned short*)(ws + 524288);               // 512 KB
    float* enorm          = (float*)(ws + 1048576);                       //   4 KB
    int*   idx_i          = (int*)  (ws + 1052672);                       // 128 KB
    float* counts         = (float*)(ws + 1183744);                       //   4 KB
    float* sse            = (float*)(ws + 1187840);                       //   4 B

    // out layout: quantized [N*D) | idx [N) | zero | commit | perplexity
    float* q_out    = out;
    float* idx_f    = out + (size_t)N_VEC * DIM;
    float* scalars  = out + (size_t)N_VEC * DIM + N_VEC;

    vq_zero_kernel<<<1, 256, 0, stream>>>(counts, sse);
    vq_prep_kernel<<<KCODE, 256, 0, stream>>>(cb, cb_hi, cb_lo, enorm);
    vq_argmin_kernel<<<N_VEC / 128, 256, 0, stream>>>(z, cb_hi, cb_lo, enorm,
                                                      idx_i, idx_f);
    vq_gather_kernel<<<N_VEC, 256, 0, stream>>>(z, cb, idx_i, q_out, counts, sse);
    vq_finalize_kernel<<<1, 256, 0, stream>>>(counts, sse, scalars);
}
